// GNN_80960133529893
// MI455X (gfx1250) — matrix operand order repaired and checked by reading, not compiled, not run on hardware
//
#include <hip/hip_runtime.h>
#include <hip/hip_bf16.h>

// ---------------------------------------------------------------------------
// 2-layer GCN for MI455X (gfx1250, wave32).
//   h1 = relu(Dm(A+I)Dm (x @ W1) + b1)        (128 -> 128)
//   h2 =      Dm(A+I)Dm (h1 @ W2) + b2        (128 -> 64)
//   out = sigmoid(mean-pool per graph)
// Dense transforms: v_wmma_f32_16x16x32_bf16 with B register-resident and A
// double-buffered (prefetched one strip ahead), guard-free (row dim padded),
// 32-bit address offsets so stores use saddr-form. Edge propagation:
// global_atomic_add_f32 into an L2-resident accumulator (51MB < 192MB L2).
// ---------------------------------------------------------------------------

typedef __attribute__((ext_vector_type(16))) __bf16 v16bf;
typedef __attribute__((ext_vector_type(8)))  float  v8f;

// ---------------- utility kernels ----------------

__global__ void fill_f32(float* __restrict__ p, float v, long n) {
  long i = (long)blockIdx.x * blockDim.x + threadIdx.x;
  if (i < n) p[i] = v;
}

__global__ void deg_accum(const long long* __restrict__ ei, float* __restrict__ deg, int E) {
  int e = blockIdx.x * blockDim.x + threadIdx.x;
  if (e < E) atomicAdd(&deg[(int)ei[E + e]], 1.0f);   // dst row of edge_index
}

__global__ void deg_to_dinv(float* __restrict__ deg, int n) {
  int i = blockIdx.x * blockDim.x + threadIdx.x;
  if (i < n) deg[i] = rsqrtf(fmaxf(deg[i], 1.0f));
}

// float -> bf16, 4 elements per thread (float4 in, 8B out)
__global__ void cast_bf16_x4(const float* __restrict__ x, __bf16* __restrict__ xb, long n4) {
  long i = (long)blockIdx.x * blockDim.x + threadIdx.x;
  if (i < n4) {
    float4 f = ((const float4*)x)[i];
    union { __bf16 h[4]; uint2 u; } o;
    o.h[0] = (__bf16)f.x; o.h[1] = (__bf16)f.y;
    o.h[2] = (__bf16)f.z; o.h[3] = (__bf16)f.w;
    ((uint2*)xb)[i] = o.u;
  }
}

// W[K][N] (row-major, f32) -> WT[N][K] (bf16): row n of WT = column n of W
__global__ void transpose_w_bf16(const float* __restrict__ W, __bf16* __restrict__ WT,
                                 int K, int N) {
  int idx = blockIdx.x * blockDim.x + threadIdx.x;
  if (idx < K * N) {
    int k = idx / N, n = idx % N;
    WT[n * K + k] = (__bf16)W[k * N + n];
  }
}

// ---------------- WMMA GEMM: C[Mpad][Nout] = A[Mpad][128] @ WT[Nout][128]^T
// Wave w owns column tile (w % NT); B fragments (16x128 of W^T) stay in
// registers. The wave grid-strides over 16-row strips with A fragments
// double-buffered: strip s+stride is prefetched before the WMMAs of strip s
// execute, so steady-state waits cover only the previous iteration's loads.
// Mpad is a multiple of 16: no guards, EXEC all-ones everywhere (WMMA req).

__global__ void wmma_gemm_bf16(const __bf16* __restrict__ A,
                               const __bf16* __restrict__ WT,
                               float* __restrict__ C,
                               int Mpad, int Nout) {
  const int K    = 128;
  const int lane = threadIdx.x & 31;
  const int half = lane >> 4;     // 0: lanes 0-15, 1: lanes 16-31
  const int l16  = lane & 15;
  const int NT   = Nout >> 4;

  const int wavesPerBlock = blockDim.x >> 5;
  const int w           = blockIdx.x * wavesPerBlock + (threadIdx.x >> 5);
  const int totalWaves  = gridDim.x * wavesPerBlock;   // multiple of NT by launch
  const int nt          = w % NT;
  const int strip0      = w / NT;
  const int stripStride = totalWaves / NT;
  const int nStrips     = Mpad >> 4;

  union Frag { v16bf v; uint4 q[2]; };

  // B fragment (32x16 bf16 per k-step): lane column = nt*16 + l16;
  // lanes 0-15 hold K[0..16), lanes 16-31 hold K[16..32) of each k-step.
  Frag bfrag[4];
  {
    unsigned boff = (unsigned)(nt * 16 + l16) * K + half * 16;
#pragma unroll
    for (int ks = 0; ks < 4; ++ks) {
      bfrag[ks].q[0] = *(const uint4*)(WT + boff + ks * 32);
      bfrag[ks].q[1] = *(const uint4*)(WT + boff + ks * 32 + 8);
    }
  }

  if (strip0 >= nStrips) return;   // wave-uniform

  // A fragment loader (16x32 bf16): lane row = l16;
  // VGPR0-7 = K[half*16..+16) (matches B) => two b128 loads.
  auto loadA = [&](Frag* a, int strip) {
    unsigned aoff = (unsigned)(strip * 16 + l16) * K + half * 16;
#pragma unroll
    for (int ks = 0; ks < 4; ++ks) {
      a[ks].q[0] = *(const uint4*)(A + aoff + ks * 32);
      a[ks].q[1] = *(const uint4*)(A + aoff + ks * 32 + 8);
    }
  };

  Frag aCur[4], aNxt[4];
  loadA(aCur, strip0);

#pragma unroll 2
  for (int strip = strip0; strip < nStrips; strip += stripStride) {
    int nxt = strip + stripStride;
    int pf  = (nxt < nStrips) ? nxt : strip;   // branchless clamp: EXEC stays full
    loadA(aNxt, pf);

    // store offsets (32-bit, uniform base) computed before the WMMA chain
    unsigned coff = (unsigned)(strip * 16 + half * 8) * Nout + nt * 16 + l16;

    v8f acc = {};
#pragma unroll
    for (int ks = 0; ks < 4; ++ks)
      acc = __builtin_amdgcn_wmma_f32_16x16x32_bf16(
          /*neg_a=*/false, aCur[ks].v, /*neg_b=*/false, bfrag[ks].v,
          /*c_mod=*/(short)0, acc, /*reuse_a=*/false, /*reuse_b=*/false);

    // C/D layout: VGPR r -> row strip*16 + half*8 + r, column nt*16 + l16
#pragma unroll
    for (int r = 0; r < 8; ++r)
      C[coff + (unsigned)r * Nout] = acc[r];

#pragma unroll
    for (int ks = 0; ks < 4; ++ks) aCur[ks] = aNxt[ks];
  }
}

// ---------------- edge scatter: agg[dst] += h[src] * dinv[src]*dinv[dst] ---
// One thread per (edge, 4-feature chunk): b128 load + 4 f32 atomics (L2).

__global__ void edge_scatter(const float* __restrict__ H,
                             const long long* __restrict__ ei,
                             const float* __restrict__ dinv,
                             float* __restrict__ agg,
                             int E, int l2F4) {       // l2F4 = log2(F/4)
  long t = (long)blockIdx.x * blockDim.x + threadIdx.x;
  if (t >= ((long)E << l2F4)) return;
  int e  = (int)(t >> l2F4);
  int f4 = (int)(t & ((1 << l2F4) - 1));
  int s = (int)ei[e];
  int d = (int)ei[E + e];
  float nrm = dinv[s] * dinv[d];
  float4 h = ((const float4*)(H + ((size_t)s << (l2F4 + 2))))[f4];
  float* od = agg + ((size_t)d << (l2F4 + 2)) + f4 * 4;
  atomicAdd(&od[0], h.x * nrm);
  atomicAdd(&od[1], h.y * nrm);
  atomicAdd(&od[2], h.z * nrm);
  atomicAdd(&od[3], h.w * nrm);
}

// ---------------- fused self-loop + bias + relu + bf16 cast ----------------

__global__ void finalize_relu_bf16(const float* __restrict__ agg,
                                   const float* __restrict__ hraw,
                                   const float* __restrict__ dinv,
                                   const float* __restrict__ bias,
                                   __bf16* __restrict__ outb,
                                   long total, int F) {
  long i = (long)blockIdx.x * blockDim.x + threadIdx.x;
  if (i >= total) return;
  int node = (int)(i / F), f = (int)(i % F);
  float di = dinv[node];
  float v = agg[i] + hraw[i] * di * di + bias[f];
  outb[i] = (__bf16)fmaxf(v, 0.0f);
}

// ---------------- layer-2 finalize + graph pooling (atomic) ----------------

__global__ void finalize_pool(const float* __restrict__ agg,
                              const float* __restrict__ hraw,
                              const float* __restrict__ dinv,
                              const float* __restrict__ bias,
                              const long long* __restrict__ batch,
                              float* __restrict__ pooled,
                              float* __restrict__ cnt,
                              long total, int F) {
  long i = (long)blockIdx.x * blockDim.x + threadIdx.x;
  if (i >= total) return;
  int node = (int)(i / F), f = (int)(i % F);
  float di = dinv[node];
  float v = agg[i] + hraw[i] * di * di + bias[f];
  int g = (int)batch[node];
  atomicAdd(&pooled[(size_t)g * F + f], v);
  if (f == 0) atomicAdd(&cnt[g], 1.0f);
}

__global__ void pool_sigmoid(const float* __restrict__ pooled,
                             const float* __restrict__ cnt,
                             float* __restrict__ out, int total, int F) {
  int i = blockIdx.x * blockDim.x + threadIdx.x;
  if (i >= total) return;
  float c = fmaxf(cnt[i / F], 1.0f);
  float v = pooled[i] / c;
  out[i] = 1.0f / (1.0f + __expf(-v));
}

// ---------------------------------------------------------------------------

extern "C" void kernel_launch(void* const* d_in, const int* in_sizes, int n_in,
                              void* d_out, int out_size, void* d_ws, size_t ws_size,
                              hipStream_t stream) {
  const float*     x     = (const float*)d_in[0];      // [N,128] f32
  const long long* ei    = (const long long*)d_in[1];  // [2,E]   i64
  const long long* batch = (const long long*)d_in[2];  // [N]     i64
  const float*     W1    = (const float*)d_in[3];      // [128,128]
  const float*     b1    = (const float*)d_in[4];      // [128]
  const float*     W2    = (const float*)d_in[5];      // [128,64]
  const float*     b2    = (const float*)d_in[6];      // [64]
  float*           out   = (float*)d_out;              // [64,64]

  const int N    = in_sizes[2];                 // 100000 nodes
  const int E    = in_sizes[1] / 2;             // 600000 edges
  const int Mpad = (N + 127) & ~127;            // padded rows: guard-free GEMM
  const int IN = 128, HID = 128, OUT = 64, G = 64;

  // workspace carving (256B aligned); big buffers sized to Mpad and reused
  auto al = [](size_t v) { return (v + 255) & ~(size_t)255; };
  char* ws = (char*)d_ws;
  size_t o = 0;
  float*  dinv   = (float*)(ws + o);  o += al((size_t)N * 4);
  __bf16* xb     = (__bf16*)(ws + o); o += al((size_t)Mpad * IN * 2);  // -> h1 bf16 later
  __bf16* w1t    = (__bf16*)(ws + o); o += al((size_t)IN * HID * 2);
  __bf16* w2t    = (__bf16*)(ws + o); o += al((size_t)HID * OUT * 2);
  float*  bufA   = (float*)(ws + o);  o += al((size_t)Mpad * HID * 4); // gemm out
  float*  bufB   = (float*)(ws + o);  o += al((size_t)Mpad * HID * 4); // scatter acc
  float*  pooled = (float*)(ws + o);  o += al((size_t)G * OUT * 4 + G * 4);
  float*  cnt    = pooled + (size_t)G * OUT;

  const int T = 256;
  auto blk = [](long n, int t) { return (unsigned)((n + t - 1) / t); };

  // --- degrees: deg = 1 (self loop) + incoming; then dinv = rsqrt(max(deg,1))
  fill_f32<<<blk(N, T), T, 0, stream>>>(dinv, 1.0f, N);
  deg_accum<<<blk(E, T), T, 0, stream>>>(ei, dinv, E);
  deg_to_dinv<<<blk(N, T), T, 0, stream>>>(dinv, N);

  // --- precision prep
  cast_bf16_x4<<<blk((long)N * IN / 4, T), T, 0, stream>>>(x, xb, (long)N * IN / 4);
  transpose_w_bf16<<<blk(IN * HID, T), T, 0, stream>>>(W1, w1t, IN, HID);
  transpose_w_bf16<<<blk(HID * OUT, T), T, 0, stream>>>(W2, w2t, HID, OUT);

  // 8 waves/block -> totalWaves = 8*grid, divisible by NT (8 and 4)
  const unsigned GEMM_GRID = 256;

  // --- layer 1: h = x@W1 ; propagate ; relu(+b1) -> bf16 (overwrites xb)
  wmma_gemm_bf16<<<GEMM_GRID, T, 0, stream>>>(xb, w1t, bufA, Mpad, HID);
  fill_f32<<<blk((long)N * HID, T), T, 0, stream>>>(bufB, 0.0f, (long)N * HID);
  edge_scatter<<<blk((long)E * (HID / 4), T), T, 0, stream>>>(bufA, ei, dinv, bufB, E, 5);
  finalize_relu_bf16<<<blk((long)N * HID, T), T, 0, stream>>>(
      bufB, bufA, dinv, b1, xb, (long)N * HID, HID);

  // --- layer 2: h = h1@W2 ; propagate ; +b2 fused into pooling
  wmma_gemm_bf16<<<GEMM_GRID, T, 0, stream>>>(xb, w2t, bufA, Mpad, OUT);
  fill_f32<<<blk((long)N * OUT, T), T, 0, stream>>>(bufB, 0.0f, (long)N * OUT);
  edge_scatter<<<blk((long)E * (OUT / 4), T), T, 0, stream>>>(bufA, ei, dinv, bufB, E, 4);

  // --- pooling + sigmoid
  fill_f32<<<blk(G * OUT + G, T), T, 0, stream>>>(pooled, 0.0f, G * OUT + G);
  finalize_pool<<<blk((long)N * OUT, T), T, 0, stream>>>(
      bufB, bufA, dinv, b2, batch, pooled, cnt, (long)N * OUT, OUT);
  pool_sigmoid<<<blk(G * OUT, T), T, 0, stream>>>(pooled, cnt, out, G * OUT, OUT);
}
